// GraphMessagePassingLayer_28398323761877
// MI455X (gfx1250) — matrix operand order repaired and checked by reading, not compiled, not run on hardware
//
#include <hip/hip_runtime.h>
#include <hip/hip_fp16.h>

typedef __attribute__((ext_vector_type(16))) _Float16 v16h;
typedef __attribute__((ext_vector_type(8)))  float    v8f;

static constexpr int B_ = 4;
static constexpr int N_ = 8192;
static constexpr int C_ = 128;
static constexpr int E_ = 131072;
static constexpr float LN_EPS = 1e-5f;

// ---------------- workspace layout (bytes) ----------------
static constexpr size_t OFF_XH   = 0;
static constexpr size_t SZ_XH    = (size_t)B_ * N_ * C_ * 2;      // 8 MB f16 x
static constexpr size_t OFF_W1T  = OFF_XH + SZ_XH;
static constexpr size_t SZ_W1T   = (size_t)128 * 256 * 2;         // msg_w1[0:256,:]^T
static constexpr size_t OFF_W2T  = OFF_W1T + SZ_W1T;
static constexpr size_t SZ_W2T   = (size_t)128 * 128 * 2;
static constexpr size_t OFF_UW1T = OFF_W2T + SZ_W2T;
static constexpr size_t SZ_UW1T  = (size_t)128 * 256 * 2;
static constexpr size_t OFF_UW2T = OFF_UW1T + SZ_UW1T;
static constexpr size_t SZ_UW2T  = (size_t)128 * 128 * 2;
static constexpr size_t OFF_AGG  = OFF_UW2T + SZ_UW2T;
static constexpr size_t SZ_AGG   = (size_t)B_ * N_ * C_ * 4;      // 16 MB f32 scatter target
static constexpr size_t OFF_ANYV = OFF_AGG + SZ_AGG;

// ---------------- dynamic LDS sizes (bytes) ----------------
// msg: Asrc(16K) + Adst(16K) + H1(16K) + W1s(64K) + W2s(32K) + meta(1K)
static constexpr size_t MSG_LDS =
    (size_t)(3 * 64 * C_ + 128 * 256 + 128 * 128) * 2 + 64 * 4 * 4;
// upd: A(32K) + H1(16K) + UW1s(64K) + UW2s(32K)
static constexpr size_t UPD_LDS =
    (size_t)(64 * 256 + 64 * C_ + 128 * 256 + 128 * 128) * 2;

// ---------------- WMMA helpers ----------------
static __device__ inline v8f wmma16x16x32(v16h a, v16h b, v8f c) {
  return __builtin_amdgcn_wmma_f32_16x16x32_f16(false, a, false, b, (short)0, c,
                                                false, false);
}

static __device__ inline v8f zero_v8f() {
  v8f z = {0.f, 0.f, 0.f, 0.f, 0.f, 0.f, 0.f, 0.f};
  return z;
}

// A-matrix 16x32 f16 fragment from an LDS tile (row-major, rl halves per row).
// ISA layout: lanes0-15 M=lane&15, K pairs {0,2,4,6,16,18,20,22}; lanes16-31 +8.
// Halves {koff..koff+7} and {koff+16..koff+23} are contiguous -> two b128 loads.
static __device__ inline v16h load_afrag(const _Float16* A, int rl, int m,
                                         int k0, int lane) {
  union { v16h v; uint4 q[2]; } a;
  const _Float16* p = A + m * rl + k0 + ((lane >> 4) << 3);
  a.q[0] = *(const uint4*)p;
  a.q[1] = *(const uint4*)(p + 16);
  return a.v;
}

// B-matrix 32x16 f16 fragment from W^T[n][k] (K halves per row), LDS or global.
// lanes0-15: col=lane, K=k0+{0..7,16..23} ; lanes16-31: col=lane-16, K=k0+{8..15,24..31}.
static __device__ inline v16h load_bfrag(const _Float16* wt, int K, int ncol,
                                         int k0, int lane) {
  union { v16h v; uint4 q[2]; } bf;
  const _Float16* p = wt + (size_t)ncol * K + k0 + ((lane >> 4) << 3);
  bf.q[0] = *(const uint4*)p;
  bf.q[1] = *(const uint4*)(p + 16);
  return bf.v;
}

// ---------------- prep kernels ----------------
__global__ void cvt_x_kernel(const float* __restrict__ x, _Float16* __restrict__ xh,
                             int n) {
  int i = blockIdx.x * blockDim.x + threadIdx.x;
  if (i < n) xh[i] = (_Float16)x[i];
}

// dst[n*K + k] = (f16) src[k*Ncol + n]   (transpose + downconvert)
__global__ void transpose_w_kernel(const float* __restrict__ src,
                                   _Float16* __restrict__ dst, int K, int Ncol) {
  int t = blockIdx.x * blockDim.x + threadIdx.x;
  if (t < K * Ncol) {
    int n = t / K;
    int k = t % K;
    dst[t] = (_Float16)src[k * Ncol + n];
  }
}

__global__ void zero_ws_kernel(float4* __restrict__ agg4, int n4,
                               int* __restrict__ anyv, int nb) {
  int i = blockIdx.x * blockDim.x + threadIdx.x;
  if (i < n4) agg4[i] = make_float4(0.f, 0.f, 0.f, 0.f);
  if (i < nb) anyv[i] = 0;
}

// ---------------- message kernel: per-edge MLP + scatter ----------------
// grid: B_*(E_/64) blocks, 128 threads (4 waves x 16 edge-rows)
__global__ __launch_bounds__(128, 1) void msg_kernel(
    const float* __restrict__ coords, const int* __restrict__ eidx,
    const unsigned char* __restrict__ evalid, const float* __restrict__ mw1,
    const float* __restrict__ mb1, const float* __restrict__ mb2,
    const _Float16* __restrict__ xh, const _Float16* __restrict__ w1t,
    const _Float16* __restrict__ w2t, float* __restrict__ agg,
    int* __restrict__ anyv) {
  extern __shared__ char smem[];
  _Float16* Asrc = (_Float16*)smem;           // 64 x 128
  _Float16* Adst = Asrc + 64 * C_;            // 64 x 128
  _Float16* H1   = Adst + 64 * C_;            // 64 x 128
  _Float16* W1s  = H1 + 64 * C_;              // 128 x 256 (w1^T)
  _Float16* W2s  = W1s + 128 * 256;           // 128 x 128 (w2^T)
  float* dxs   = (float*)(W2s + 128 * 128);
  float* dys   = dxs + 64;
  float* maskf = dys + 64;
  int*   dsts  = (int*)(maskf + 64);

  const int tid = threadIdx.x;
  const int lane = tid & 31;
  const int wave = tid >> 5;
  const int b = blockIdx.x / (E_ / 64);
  const int e0 = (blockIdx.x % (E_ / 64)) * 64;

  // ---- stage block-invariant weights into LDS (96 KB) ----
  {
    const uint4* g1 = (const uint4*)w1t;  // 4096 uint4
    uint4* s1 = (uint4*)W1s;
#pragma unroll
    for (int i = 0; i < 32; ++i) s1[tid + 128 * i] = g1[tid + 128 * i];
    const uint4* g2 = (const uint4*)w2t;  // 2048 uint4
    uint4* s2 = (uint4*)W2s;
#pragma unroll
    for (int i = 0; i < 16; ++i) s2[tid + 128 * i] = g2[tid + 128 * i];
  }

  // ---- cooperative gather of x_src / x_dst rows (f16) + edge metadata ----
  {
    const int r = tid & 63;
    const int part = tid >> 6;  // which 64-half chunk of the row
    const int e = e0 + r;
    const int si = eidx[e];
    const int di = eidx[E_ + e];
    const uint4* sp = (const uint4*)(xh + ((size_t)b * N_ + si) * C_ + part * 64);
    const uint4* dp = (const uint4*)(xh + ((size_t)b * N_ + di) * C_ + part * 64);
    uint4* sd = (uint4*)(Asrc + r * C_ + part * 64);
    uint4* dd = (uint4*)(Adst + r * C_ + part * 64);
#pragma unroll
    for (int i = 0; i < 8; ++i) {
      sd[i] = sp[i];
      dd[i] = dp[i];
    }
    if (part == 0) {
      dsts[r] = di;
      dxs[r] = coords[((size_t)b * N_ + di) * 2 + 0] -
               coords[((size_t)b * N_ + si) * 2 + 0];
      dys[r] = coords[((size_t)b * N_ + di) * 2 + 1] -
               coords[((size_t)b * N_ + si) * 2 + 1];
      maskf[r] = evalid[(size_t)b * E_ + e] ? 1.0f : 0.0f;
    }
  }
  __syncthreads();

  const int m0 = wave * 16;
  const int mrow = m0 + (lane & 15);
  const int colq = lane & 15;

  // ---- GEMM1: [x_src | x_dst] (64x256 f16) @ w1t -> h1 (64x128 f32) ----
  v8f acc[8];
#pragma unroll
  for (int nt = 0; nt < 8; ++nt) acc[nt] = zero_v8f();

#pragma unroll
  for (int ks = 0; ks < 8; ++ks) {
    const v16h a = (ks < 4) ? load_afrag(Asrc, C_, mrow, ks * 32, lane)
                            : load_afrag(Adst, C_, mrow, (ks - 4) * 32, lane);
    v16h bfr[8];
#pragma unroll
    for (int nt = 0; nt < 8; ++nt)
      bfr[nt] = load_bfrag(W1s, 256, nt * 16 + colq, ks * 32, lane);
#pragma unroll
    for (int nt = 0; nt < 8; ++nt) acc[nt] = wmma16x16x32(a, bfr[nt], acc[nt]);
  }

  // ---- epilogue1: + bias + rank-2 delta term, ReLU, to LDS f16 ----
#pragma unroll
  for (int nt = 0; nt < 8; ++nt) {
    const int ncol = nt * 16 + colq;
    const float w256 = mw1[256 * C_ + ncol];
    const float w257 = mw1[257 * C_ + ncol];
    const float bias = mb1[ncol];
#pragma unroll
    for (int v = 0; v < 8; ++v) {
      const int row = m0 + v + ((lane >> 4) << 3);
      float val = acc[nt][v] + bias + dxs[row] * w256 + dys[row] * w257;
      H1[row * C_ + ncol] = (_Float16)fmaxf(val, 0.0f);
    }
  }
  __syncthreads();

  // ---- GEMM2: h1 (64x128 f16) @ w2t -> msg (64x128 f32) ----
  v8f acc2[8];
#pragma unroll
  for (int nt = 0; nt < 8; ++nt) acc2[nt] = zero_v8f();
#pragma unroll
  for (int ks = 0; ks < 4; ++ks) {
    const v16h a = load_afrag(H1, C_, mrow, ks * 32, lane);
    v16h bfr[8];
#pragma unroll
    for (int nt = 0; nt < 8; ++nt)
      bfr[nt] = load_bfrag(W2s, 128, nt * 16 + colq, ks * 32, lane);
#pragma unroll
    for (int nt = 0; nt < 8; ++nt) acc2[nt] = wmma16x16x32(a, bfr[nt], acc2[nt]);
  }

  // ---- mask + scatter-add into agg[b][dst] ----
#pragma unroll
  for (int nt = 0; nt < 8; ++nt) {
    const int ncol = nt * 16 + colq;
    const float bias = mb2[ncol];
#pragma unroll
    for (int v = 0; v < 8; ++v) {
      const int row = m0 + v + ((lane >> 4) << 3);
      const float val = (acc2[nt][v] + bias) * maskf[row];
      atomicAdd(&agg[((size_t)b * N_ + dsts[row]) * C_ + ncol], val);
    }
  }
  if (tid < 64 && maskf[tid] != 0.0f) anyv[b] = 1;
}

// ---------------- update kernel: per-node MLP + residual + LN ----------------
// grid: B_*(N_/64) blocks, 128 threads (4 waves x 16 node-rows)
__global__ __launch_bounds__(128, 1) void upd_kernel(
    const float* __restrict__ x, const float* __restrict__ ub1,
    const float* __restrict__ ub2, const float* __restrict__ lng,
    const float* __restrict__ lnb, const _Float16* __restrict__ xh,
    const float* __restrict__ agg, const _Float16* __restrict__ uw1t,
    const _Float16* __restrict__ uw2t, const int* __restrict__ anyv,
    float* __restrict__ out) {
  extern __shared__ char smem[];
  _Float16* A    = (_Float16*)smem;        // 64 x 256  [x | agg]
  _Float16* H1   = A + 64 * 256;           // 64 x 128
  _Float16* W1s  = H1 + 64 * C_;           // 128 x 256
  _Float16* W2s  = W1s + 128 * 256;        // 128 x 128

  const int tid = threadIdx.x;
  const int lane = tid & 31;
  const int wave = tid >> 5;
  const int b = blockIdx.x / (N_ / 64);
  const int n0 = (blockIdx.x % (N_ / 64)) * 64;

  // ---- stage weights into LDS ----
  {
    const uint4* g1 = (const uint4*)uw1t;
    uint4* s1 = (uint4*)W1s;
#pragma unroll
    for (int i = 0; i < 32; ++i) s1[tid + 128 * i] = g1[tid + 128 * i];
    const uint4* g2 = (const uint4*)uw2t;
    uint4* s2 = (uint4*)W2s;
#pragma unroll
    for (int i = 0; i < 16; ++i) s2[tid + 128 * i] = g2[tid + 128 * i];
  }

  // ---- stage [x | agg] rows into LDS as f16 ----
  {
    const int r = tid & 63;
    const int part = tid >> 6;
    const int node = n0 + r;
    const uint4* sp = (const uint4*)(xh + ((size_t)b * N_ + node) * C_ + part * 64);
    uint4* dd = (uint4*)(A + r * 256 + part * 64);
#pragma unroll
    for (int i = 0; i < 8; ++i) dd[i] = sp[i];
    const float4* ap =
        (const float4*)(agg + ((size_t)b * N_ + node) * C_ + part * 64);
#pragma unroll
    for (int i = 0; i < 16; ++i) {
      float4 f = ap[i];
      const int c0 = C_ + part * 64 + i * 4;
      A[r * 256 + c0 + 0] = (_Float16)f.x;
      A[r * 256 + c0 + 1] = (_Float16)f.y;
      A[r * 256 + c0 + 2] = (_Float16)f.z;
      A[r * 256 + c0 + 3] = (_Float16)f.w;
    }
  }
  __syncthreads();

  const int m0 = wave * 16;
  const int mrow = m0 + (lane & 15);
  const int colq = lane & 15;

  // ---- GEMM1: [x|agg] (64x256) @ uw1t ----
  v8f acc[8];
#pragma unroll
  for (int nt = 0; nt < 8; ++nt) acc[nt] = zero_v8f();
#pragma unroll
  for (int ks = 0; ks < 8; ++ks) {
    const v16h a = load_afrag(A, 256, mrow, ks * 32, lane);
    v16h bfr[8];
#pragma unroll
    for (int nt = 0; nt < 8; ++nt)
      bfr[nt] = load_bfrag(W1s, 256, nt * 16 + colq, ks * 32, lane);
#pragma unroll
    for (int nt = 0; nt < 8; ++nt) acc[nt] = wmma16x16x32(a, bfr[nt], acc[nt]);
  }

#pragma unroll
  for (int nt = 0; nt < 8; ++nt) {
    const int ncol = nt * 16 + colq;
    const float bias = ub1[ncol];
#pragma unroll
    for (int v = 0; v < 8; ++v) {
      const int row = m0 + v + ((lane >> 4) << 3);
      H1[row * C_ + ncol] = (_Float16)fmaxf(acc[nt][v] + bias, 0.0f);
    }
  }
  __syncthreads();

  // ---- GEMM2: h1 @ uw2t -> upd ----
  v8f acc2[8];
#pragma unroll
  for (int nt = 0; nt < 8; ++nt) acc2[nt] = zero_v8f();
#pragma unroll
  for (int ks = 0; ks < 4; ++ks) {
    const v16h a = load_afrag(H1, C_, mrow, ks * 32, lane);
    v16h bfr[8];
#pragma unroll
    for (int nt = 0; nt < 8; ++nt)
      bfr[nt] = load_bfrag(W2s, 128, nt * 16 + colq, ks * 32, lane);
#pragma unroll
    for (int nt = 0; nt < 8; ++nt) acc2[nt] = wmma16x16x32(a, bfr[nt], acc2[nt]);
  }

  // ---- residual + LayerNorm; row m lives on 16 lanes of one half-wave ----
  const int flag = anyv[b];
  float y[8][8];
  float s1[8], s2[8];
#pragma unroll
  for (int v = 0; v < 8; ++v) { s1[v] = 0.f; s2[v] = 0.f; }

#pragma unroll
  for (int nt = 0; nt < 8; ++nt) {
    const int ncol = nt * 16 + colq;
    const float bias = ub2[ncol];
#pragma unroll
    for (int v = 0; v < 8; ++v) {
      const int row = m0 + v + ((lane >> 4) << 3);
      const float xv = x[((size_t)b * N_ + n0 + row) * C_ + ncol];
      const float yv = xv + acc2[nt][v] + bias;
      y[nt][v] = yv;
      s1[v] += yv;
      s2[v] += yv * yv;
    }
  }
  // reduce over the 16 lanes of this half-wave (xor 1,2,4,8 stays in-half)
#pragma unroll
  for (int v = 0; v < 8; ++v) {
#pragma unroll
    for (int m = 1; m <= 8; m <<= 1) {
      s1[v] += __shfl_xor(s1[v], m, 32);
      s2[v] += __shfl_xor(s2[v], m, 32);
    }
  }
  float mu[8], rs[8];
#pragma unroll
  for (int v = 0; v < 8; ++v) {
    mu[v] = s1[v] * (1.0f / C_);
    const float var = s2[v] * (1.0f / C_) - mu[v] * mu[v];
    rs[v] = __frsqrt_rn(var + LN_EPS);
  }

#pragma unroll
  for (int nt = 0; nt < 8; ++nt) {
    const int ncol = nt * 16 + colq;
    const float g = lng[ncol];
    const float bb = lnb[ncol];
#pragma unroll
    for (int v = 0; v < 8; ++v) {
      const int row = m0 + v + ((lane >> 4) << 3);
      const size_t o = ((size_t)b * N_ + n0 + row) * C_ + ncol;
      const float yn = (y[nt][v] - mu[v]) * rs[v] * g + bb;
      out[o] = flag ? yn : x[o];
    }
  }
}

// ---------------- host launcher ----------------
extern "C" void kernel_launch(void* const* d_in, const int* in_sizes, int n_in,
                              void* d_out, int out_size, void* d_ws,
                              size_t ws_size, hipStream_t stream) {
  (void)in_sizes; (void)n_in; (void)out_size; (void)ws_size;
  const float* x = (const float*)d_in[0];
  const float* coords = (const float*)d_in[1];
  const int* eidx = (const int*)d_in[2];
  const unsigned char* evalid = (const unsigned char*)d_in[3];
  const float* mw1 = (const float*)d_in[4];
  const float* mb1 = (const float*)d_in[5];
  const float* mw2 = (const float*)d_in[6];
  const float* mb2 = (const float*)d_in[7];
  const float* uw1 = (const float*)d_in[8];
  const float* ub1 = (const float*)d_in[9];
  const float* uw2 = (const float*)d_in[10];
  const float* ub2 = (const float*)d_in[11];
  const float* lng = (const float*)d_in[12];
  const float* lnb = (const float*)d_in[13];

  char* ws = (char*)d_ws;
  _Float16* xh = (_Float16*)(ws + OFF_XH);
  _Float16* w1t = (_Float16*)(ws + OFF_W1T);
  _Float16* w2t = (_Float16*)(ws + OFF_W2T);
  _Float16* uw1t = (_Float16*)(ws + OFF_UW1T);
  _Float16* uw2t = (_Float16*)(ws + OFF_UW2T);
  float* agg = (float*)(ws + OFF_AGG);
  int* anyv = (int*)(ws + OFF_ANYV);
  float* out = (float*)d_out;

  // allow >64KB dynamic LDS (gfx1250 WGP has 320KB); host-side, capture-safe
  (void)hipFuncSetAttribute((const void*)msg_kernel,
                            hipFuncAttributeMaxDynamicSharedMemorySize,
                            (int)MSG_LDS);
  (void)hipFuncSetAttribute((const void*)upd_kernel,
                            hipFuncAttributeMaxDynamicSharedMemorySize,
                            (int)UPD_LDS);

  const int nx = B_ * N_ * C_;
  cvt_x_kernel<<<(nx + 255) / 256, 256, 0, stream>>>(x, xh, nx);
  transpose_w_kernel<<<(256 * 128 + 255) / 256, 256, 0, stream>>>(mw1, w1t, 256, 128);
  transpose_w_kernel<<<(128 * 128 + 255) / 256, 256, 0, stream>>>(mw2, w2t, 128, 128);
  transpose_w_kernel<<<(256 * 128 + 255) / 256, 256, 0, stream>>>(uw1, uw1t, 256, 128);
  transpose_w_kernel<<<(128 * 128 + 255) / 256, 256, 0, stream>>>(uw2, uw2t, 128, 128);
  zero_ws_kernel<<<(nx / 4 + 255) / 256, 256, 0, stream>>>((float4*)agg, nx / 4,
                                                           anyv, B_);
  msg_kernel<<<B_ * (E_ / 64), 128, MSG_LDS, stream>>>(
      coords, eidx, evalid, mw1, mb1, mb2, xh, w1t, w2t, agg, anyv);
  upd_kernel<<<B_ * (N_ / 64), 128, UPD_LDS, stream>>>(
      x, ub1, ub2, lng, lnb, xh, agg, uw1t, uw2t, anyv, out);
}
